// EfficientViTJointBlock_60318520705753
// MI455X (gfx1250) — matrix operand order repaired and checked by reading, not compiled, not run on hardware
//
#include <hip/hip_runtime.h>
#include <cstdint>

// ---------------------------------------------------------------------------
// Types
// ---------------------------------------------------------------------------
typedef __bf16 bf16;
typedef __attribute__((ext_vector_type(16))) __bf16 v16bf;
typedef __attribute__((ext_vector_type(8)))  __bf16 v8bf;
typedef __attribute__((ext_vector_type(8)))  float   v8f;
typedef __attribute__((ext_vector_type(4)))  uint32_t v4u;
typedef __attribute__((ext_vector_type(8)))  int      v8i;
typedef __attribute__((ext_vector_type(4)))  int      v4i;

// Problem constants (from reference)
constexpr int Cc    = 256;
constexpr int Aa    = 256;
constexpr int NH    = 8;
constexpr int HD    = 32;
constexpr int HID   = 1024;
constexpr int BATCH = 8;
constexpr int Himg  = 96;
constexpr int Wimg  = 96;
constexpr int HW_   = Himg * Wimg;      // 9216

// ---------------------------------------------------------------------------
// TDM: 2D tile load (64 rows x 32 bf16) from row-major [*, K] matrix into LDS.
// D# per CDNA5 ISA ch.8. LDS padding: pad_interval=16 dwords (code 3),
// pad_amount=4 dwords (code 3) -> LDS row pitch = 20 dwords = 40 bf16.
// 6-arg builtin: (u32x4 g0, i32x8 g1, i32x4 g2, i32x4 g3, i32x8, i32 cpol)
// ---------------------------------------------------------------------------
__device__ inline void tdm_load_tile_64x32(uint32_t lds_off, const bf16* gptr,
                                           int K, int Mrows) {
  uint64_t ga = (uint64_t)(uintptr_t)gptr;
  v4u g0;
  g0[0] = 1u;                                          // count=1, user D#
  g0[1] = lds_off;                                     // lds_addr (bytes)
  g0[2] = (uint32_t)(ga & 0xFFFFFFFFu);                // global_addr lo
  g0[3] = (uint32_t)((ga >> 32) & 0x1FFFFFFu) | (2u << 30); // hi25 | type=2
  v8i g1;
  g1[0] = (int)((1u << 16) |      // data_size = 2 bytes
                (1u << 20) |      // pad_enable
                (3u << 22) |      // pad_interval: 16 dwords
                (3u << 25));      // pad_amount: 4 dwords
  g1[1] = (int)(((uint32_t)K & 0xFFFFu) << 16);        // tensor_dim0 lo16
  g1[2] = (int)((((uint32_t)K >> 16) & 0xFFFFu) |
                (((uint32_t)Mrows & 0xFFFFu) << 16));  // dim0 hi | dim1 lo
  g1[3] = (int)(((((uint32_t)Mrows) >> 16) & 0xFFFFu) |
                (32u << 16));                          // dim1 hi | tile_dim0=32
  g1[4] = (int)64u;                                    // tile_dim1=64, tile_dim2=0
  g1[5] = (int)K;                                      // tensor_dim0_stride lo32
  g1[6] = 0;
  g1[7] = 0;
  v4i z4 = {0, 0, 0, 0};
  v8i z8 = {0, 0, 0, 0, 0, 0, 0, 0};
  __builtin_amdgcn_tensor_load_to_lds(g0, g1, z4, z4, z8, 0);
}

// ---------------------------------------------------------------------------
// WMMA GEMM for pointwise conv: Out[b][M][HW] = W[M][K] * In[b][K][HW] (+epi)
// MODE 0: out(bf16) = acc + bias
// MODE 1: out(bf16) = acc + bias + extra(bf16)
// MODE 2: out(f32)  = acc + bias + extra(f32)
// Block: 256 thr (8 waves), tile 64(M) x 64(N), K-step 32.
// ---------------------------------------------------------------------------
template <int MODE>
__global__ __launch_bounds__(256) void gemm_pw_kernel(
    const bf16* __restrict__ Wb, const float* __restrict__ bias,
    const bf16* __restrict__ In, const void* __restrict__ extra,
    void* __restrict__ Out, int K, int M) {
  __shared__ bf16 lds_A[64][40];   // M x K tile (40 = 20-dword TDM pitch)
  __shared__ bf16 lds_B[64][40];   // N x K tile (transposed activations)

  const int tid  = threadIdx.x;
  const int p0   = blockIdx.x * 64;
  const int o0   = blockIdx.y * 64;
  const int bidx = blockIdx.z;
  const size_t inBase  = (size_t)bidx * K * HW_;
  const size_t outBase = (size_t)bidx * M * HW_;
  const int lane = tid & 31, wave = tid >> 5;
  const int mw = wave & 3, nw = wave >> 2;     // 4x2 wave grid
  const int half = lane >> 4, l16 = lane & 15;

  v8f acc0 = {};
  v8f acc1 = {};

  // B staging: one b128 per thread (8 contiguous pixels of one k-row),
  // scattered into the transposed LDS tile.
  const int kStage = tid >> 3;         // 0..31
  const int pStage = (tid & 7) * 8;    // 0,8,...,56
  const uint32_t ldsAoff = (uint32_t)(uintptr_t)(&lds_A[0][0]);

  for (int k0 = 0; k0 < K; k0 += 32) {
    __syncthreads();
    // A tile via Tensor Data Mover (issued by wave 0 only; EXEC ignored)
    if (wave == 0)
      tdm_load_tile_64x32(ldsAoff, Wb + (size_t)o0 * K + k0, K, M);
    // B tile staged transposed: lds_B[n][k] = In[k0+k][p0+n]
    {
      const v8bf bv = *(const v8bf*)&In[inBase + (size_t)(k0 + kStage) * HW_ +
                                        p0 + pStage];
      #pragma unroll
      for (int j = 0; j < 8; ++j) lds_B[pStage + j][kStage] = bv[j];
    }
    if (wave == 0) __builtin_amdgcn_s_wait_tensorcnt(0);
    __syncthreads();

    // A fragment (16x32 bf16): lane holds row m; k chunk at 16*half
    v16bf a;
    {
      const v8bf lo = *(const v8bf*)&lds_A[mw * 16 + l16][16 * half];
      const v8bf hi = *(const v8bf*)&lds_A[mw * 16 + l16][16 * half + 8];
      #pragma unroll
      for (int i = 0; i < 8; ++i) { a[i] = lo[i]; a[i + 8] = hi[i]; }
    }
    // B fragments (32x16 bf16): lane holds column n; k chunk at 16*half
    v16bf b0, b1;
    {
      const v8bf x0 = *(const v8bf*)&lds_B[nw * 32 + l16][16 * half];
      const v8bf x1 = *(const v8bf*)&lds_B[nw * 32 + l16][16 * half + 8];
      const v8bf y0 = *(const v8bf*)&lds_B[nw * 32 + 16 + l16][16 * half];
      const v8bf y1 = *(const v8bf*)&lds_B[nw * 32 + 16 + l16][16 * half + 8];
      #pragma unroll
      for (int i = 0; i < 8; ++i) {
        b0[i] = x0[i]; b0[i + 8] = x1[i];
        b1[i] = y0[i]; b1[i + 8] = y1[i];
      }
    }
    acc0 = __builtin_amdgcn_wmma_f32_16x16x32_bf16(false, a, false, b0,
                                                   (short)0, acc0, false, false);
    acc1 = __builtin_amdgcn_wmma_f32_16x16x32_bf16(false, a, false, b1,
                                                   (short)0, acc1, false, false);
  }

  // Epilogue. D layout: VGPR r -> M = r + 8*half, N = lane&15
  #pragma unroll
  for (int r = 0; r < 8; ++r) {
    const int row = o0 + mw * 16 + r + 8 * half;
    const float bs = bias[row];
    const size_t i0 = outBase + (size_t)row * HW_ + p0 + nw * 32 + l16;
    float v0 = acc0[r] + bs;
    float v1 = acc1[r] + bs;
    if (MODE == 0) {
      ((bf16*)Out)[i0]      = (bf16)v0;
      ((bf16*)Out)[i0 + 16] = (bf16)v1;
    } else if (MODE == 1) {
      v0 += (float)((const bf16*)extra)[i0];
      v1 += (float)((const bf16*)extra)[i0 + 16];
      ((bf16*)Out)[i0]      = (bf16)v0;
      ((bf16*)Out)[i0 + 16] = (bf16)v1;
    } else {
      v0 += ((const float*)extra)[i0];
      v1 += ((const float*)extra)[i0 + 16];
      ((float*)Out)[i0]      = v0;
      ((float*)Out)[i0 + 16] = v1;
    }
  }
}

// ---------------------------------------------------------------------------
// LayerNorm over channel dim of [B,Cn,HW] f32 -> bf16
// ---------------------------------------------------------------------------
__global__ __launch_bounds__(256) void ln_kernel(
    const float* __restrict__ x, const float* __restrict__ g,
    const float* __restrict__ bb, bf16* __restrict__ out, int Cn) {
  const int p = blockIdx.x * 256 + threadIdx.x;
  const int b = blockIdx.y;
  const size_t base = (size_t)b * Cn * HW_ + p;
  float s = 0.f, s2 = 0.f;
  for (int c = 0; c < Cn; ++c) {
    const float v = x[base + (size_t)c * HW_];
    s += v; s2 += v * v;
  }
  const float mu  = s / (float)Cn;
  const float var = s2 / (float)Cn - mu * mu;
  const float rs  = rsqrtf(var + 1e-5f);
  for (int c = 0; c < Cn; ++c) {
    const float v = x[base + (size_t)c * HW_];
    out[base + (size_t)c * HW_] = (bf16)((v - mu) * rs * g[c] + bb[c]);
  }
}

// ---------------------------------------------------------------------------
// Depthwise conv: dw3 (+ optional dw5) with biases, optional exact GELU.
// chanStride = channels per batch in the buffer (supports qkv sub-views).
// ---------------------------------------------------------------------------
template <bool HAS5, bool GELU>
__global__ __launch_bounds__(256) void dw_kernel(
    const bf16* __restrict__ in, bf16* __restrict__ out,
    const float* __restrict__ w3, const float* __restrict__ b3,
    const float* __restrict__ w5, const float* __restrict__ b5,
    int chanStride) {
  const int p = blockIdx.x * 256 + threadIdx.x;
  const int c = blockIdx.y;
  const int b = blockIdx.z;
  const int py = p / Wimg, px = p % Wimg;
  const size_t base = ((size_t)b * chanStride + c) * HW_;
  float acc = b3[c] + (HAS5 ? b5[c] : 0.f);
  #pragma unroll
  for (int dy = -1; dy <= 1; ++dy)
    #pragma unroll
    for (int dx = -1; dx <= 1; ++dx) {
      const int yy = py + dy, xx = px + dx;
      if (yy >= 0 && yy < Himg && xx >= 0 && xx < Wimg)
        acc += w3[c * 9 + (dy + 1) * 3 + (dx + 1)] *
               (float)in[base + (size_t)yy * Wimg + xx];
    }
  if (HAS5) {
    #pragma unroll
    for (int dy = -2; dy <= 2; ++dy)
      #pragma unroll
      for (int dx = -2; dx <= 2; ++dx) {
        const int yy = py + dy, xx = px + dx;
        if (yy >= 0 && yy < Himg && xx >= 0 && xx < Wimg)
          acc += w5[c * 25 + (dy + 2) * 5 + (dx + 2)] *
                 (float)in[base + (size_t)yy * Wimg + xx];
      }
  }
  if (GELU) acc = 0.5f * acc * (1.f + erff(acc * 0.70710678118654752440f));
  out[base + p] = (bf16)acc;
}

// ---------------------------------------------------------------------------
// Linear attention stats: per (b,h) compute kv[32][32] = relu(K)^T V and
// ksum[32]. qkv buffer layout: channels [0,256)=q, [256,512)=k, [512,768)=v.
// ---------------------------------------------------------------------------
__global__ __launch_bounds__(256) void attn_stats_kernel(
    const bf16* __restrict__ qkv, float* __restrict__ kv,
    float* __restrict__ ksum) {
  __shared__ bf16 Kt[32][132];
  __shared__ bf16 Vt[32][132];
  const int bh = blockIdx.x;
  const int b = bh >> 3, h = bh & 7;
  const bf16* kp = qkv + ((size_t)b * 768 + 256 + h * 32) * HW_;
  const bf16* vp = qkv + ((size_t)b * 768 + 512 + h * 32) * HW_;
  const int tid = threadIdx.x;
  const int d = tid >> 3, e0 = (tid & 7) * 4;
  const int rS = tid >> 3, j0 = (tid & 7) * 16;
  float a0 = 0.f, a1 = 0.f, a2 = 0.f, a3 = 0.f, ks = 0.f;
  for (int n0 = 0; n0 < HW_; n0 += 128) {
    __syncthreads();
    #pragma unroll
    for (int i = 0; i < 16; ++i) {
      const float kl = (float)kp[(size_t)rS * HW_ + n0 + j0 + i];
      Kt[rS][j0 + i] = (bf16)(kl > 0.f ? kl : 0.f);   // relu(k)
      Vt[rS][j0 + i] = vp[(size_t)rS * HW_ + n0 + j0 + i];
    }
    __syncthreads();
    for (int j = 0; j < 128; ++j) {
      const float kd = (float)Kt[d][j];
      a0 += kd * (float)Vt[e0 + 0][j];
      a1 += kd * (float)Vt[e0 + 1][j];
      a2 += kd * (float)Vt[e0 + 2][j];
      a3 += kd * (float)Vt[e0 + 3][j];
    }
    if (tid < 32)
      for (int j = 0; j < 128; ++j) ks += (float)Kt[tid][j];
  }
  float* kvb = kv + (size_t)bh * 1024;
  kvb[d * 32 + e0 + 0] = a0;
  kvb[d * 32 + e0 + 1] = a1;
  kvb[d * 32 + e0 + 2] = a2;
  kvb[d * 32 + e0 + 3] = a3;
  if (tid < 32) ksum[bh * 32 + tid] = ks;
}

// ---------------------------------------------------------------------------
// Linear attention apply: out[b, h*32+e, p] = relu(q).kv[:,e] / max(q.ksum,eps)
// ---------------------------------------------------------------------------
__global__ __launch_bounds__(256) void attn_apply_kernel(
    const bf16* __restrict__ qkv, const float* __restrict__ kv,
    const float* __restrict__ ksum, bf16* __restrict__ out) {
  __shared__ float skv[1024];
  __shared__ float sks[32];
  const int tid = threadIdx.x;
  const int h = blockIdx.y, b = blockIdx.z;
  const int bh = b * 8 + h;
  #pragma unroll
  for (int i = 0; i < 4; ++i)
    skv[tid + 256 * i] = kv[(size_t)bh * 1024 + tid + 256 * i];
  if (tid < 32) sks[tid] = ksum[bh * 32 + tid];
  __syncthreads();
  const int p = blockIdx.x * 256 + tid;
  const bf16* qp = qkv + ((size_t)b * 768 + h * 32) * HW_;
  float q[32];
  float den = 0.f;
  #pragma unroll
  for (int d = 0; d < 32; ++d) {
    float v = (float)qp[(size_t)d * HW_ + p];
    v = v > 0.f ? v : 0.f;
    q[d] = v;
    den += v * sks[d];
  }
  den = fmaxf(den, 1e-6f);
  const float inv = 1.f / den;
  bf16* op = out + ((size_t)b * 256 + h * 32) * HW_ + p;
  #pragma unroll
  for (int e = 0; e < 32; ++e) {
    float num = 0.f;
    #pragma unroll
    for (int d = 0; d < 32; ++d) num += q[d] * skv[d * 32 + e];
    op[(size_t)e * HW_] = (bf16)(num * inv);
  }
}

// f32 -> bf16 conversion
__global__ __launch_bounds__(256) void cvt_kernel(const float* __restrict__ s,
                                                  bf16* __restrict__ d, int n) {
  const int i = blockIdx.x * 256 + threadIdx.x;
  if (i < n) d[i] = (bf16)s[i];
}

// ---------------------------------------------------------------------------
// Host launch
// ---------------------------------------------------------------------------
extern "C" void kernel_launch(void* const* d_in, const int* in_sizes, int n_in,
                              void* d_out, int out_size, void* d_ws,
                              size_t ws_size, hipStream_t stream) {
  (void)in_sizes; (void)n_in; (void)out_size; (void)ws_size;
  const float* x            = (const float*)d_in[0];
  const float* pre_g        = (const float*)d_in[1];
  const float* pre_b        = (const float*)d_in[2];
  const float* local_pw_w   = (const float*)d_in[3];
  const float* local_pw_b   = (const float*)d_in[4];
  const float* local_dw3_w  = (const float*)d_in[5];
  const float* local_dw3_b  = (const float*)d_in[6];
  const float* local_dw5_w  = (const float*)d_in[7];
  const float* local_dw5_b  = (const float*)d_in[8];
  const float* local_fuse_w = (const float*)d_in[9];
  const float* local_fuse_b = (const float*)d_in[10];
  const float* qkv_w        = (const float*)d_in[11];
  const float* qkv_b        = (const float*)d_in[12];
  const float* q_dw3_w      = (const float*)d_in[13];
  const float* q_dw3_b      = (const float*)d_in[14];
  const float* q_dw5_w      = (const float*)d_in[15];
  const float* q_dw5_b      = (const float*)d_in[16];
  const float* k_dw3_w      = (const float*)d_in[17];
  const float* k_dw3_b      = (const float*)d_in[18];
  const float* k_dw5_w      = (const float*)d_in[19];
  const float* k_dw5_b      = (const float*)d_in[20];
  const float* v_dw3_w      = (const float*)d_in[21];
  const float* v_dw3_b      = (const float*)d_in[22];
  const float* v_dw5_w      = (const float*)d_in[23];
  const float* v_dw5_b      = (const float*)d_in[24];
  const float* global_out_w = (const float*)d_in[25];
  const float* global_out_b = (const float*)d_in[26];
  const float* fuse_pw_w    = (const float*)d_in[27];
  const float* fuse_pw_b    = (const float*)d_in[28];
  const float* ffn_g        = (const float*)d_in[29];
  const float* ffn_b        = (const float*)d_in[30];
  const float* ffn_exp_w    = (const float*)d_in[31];
  const float* ffn_exp_b    = (const float*)d_in[32];
  const float* ffn_dw3_w    = (const float*)d_in[33];
  const float* ffn_dw3_b    = (const float*)d_in[34];
  const float* ffn_shr_w    = (const float*)d_in[35];
  const float* ffn_shr_b    = (const float*)d_in[36];

  char* ws = (char*)d_ws;
  const size_t MB = (size_t)1 << 20;
  // bf16 weight pool
  bf16* wb           = (bf16*)ws;
  bf16* wb_local_pw  = wb + 0;
  bf16* wb_local_fu  = wb + 65536;
  bf16* wb_qkv       = wb + 131072;
  bf16* wb_global    = wb + 327680;
  bf16* wb_fuse      = wb + 393216;
  bf16* wb_exp       = wb + 458752;
  bf16* wb_shr       = wb + 720896;
  float* kvbuf = (float*)(ws + 2 * MB);
  float* ksbuf = (float*)(ws + 2 * MB + 512 * 1024);
  bf16* bufA  = (bf16*)(ws + 4 * MB);     // xn, later (local+global) sum
  bf16* bufB  = (bf16*)(ws + 40 * MB);    // seed / localf / ffn-normed
  bf16* bufC  = (bf16*)(ws + 76 * MB);    // local / attn out
  bf16* bufD  = (bf16*)(ws + 112 * MB);   // qkv raw, later y1 (f32)
  bf16* bufE  = (bf16*)(ws + 220 * MB);   // qkv after dw
  bf16* bufF1 = (bf16*)(ws + 328 * MB);   // ffn expand
  bf16* bufF2 = (bf16*)(ws + 472 * MB);   // ffn after dw3+gelu
  float* y1   = (float*)(ws + 112 * MB);

  const dim3 blk(256);
  auto cvt = [&](const float* s, bf16* d, int n) {
    cvt_kernel<<<dim3((n + 255) / 256), blk, 0, stream>>>(s, d, n);
  };
  cvt(local_pw_w,   wb_local_pw, 256 * 256);
  cvt(local_fuse_w, wb_local_fu, 256 * 256);
  cvt(qkv_w,        wb_qkv,      768 * 256);
  cvt(global_out_w, wb_global,   256 * 256);
  cvt(fuse_pw_w,    wb_fuse,     256 * 256);
  cvt(ffn_exp_w,    wb_exp,      1024 * 256);
  cvt(ffn_shr_w,    wb_shr,      256 * 1024);

  const dim3 gLN(HW_ / 256, BATCH);
  const dim3 gDW(HW_ / 256, 256, BATCH);
  const dim3 gAP(HW_ / 256, NH, BATCH);

  // 1) pre-norm
  ln_kernel<<<gLN, blk, 0, stream>>>(x, pre_g, pre_b, bufA, Cc);
  // 2) local seed = local_pw(xn)
  gemm_pw_kernel<0><<<dim3(144, 4, BATCH), blk, 0, stream>>>(
      wb_local_pw, local_pw_b, bufA, nullptr, bufB, 256, 256);
  // 3) local = dw3(seed) + dw5(seed)
  dw_kernel<true, false><<<gDW, blk, 0, stream>>>(
      bufB, bufC, local_dw3_w, local_dw3_b, local_dw5_w, local_dw5_b, 256);
  // 4) localf = local_fuse(local)
  gemm_pw_kernel<0><<<dim3(144, 4, BATCH), blk, 0, stream>>>(
      wb_local_fu, local_fuse_b, bufC, nullptr, bufB, 256, 256);
  // 5) qkv = qkv_pw(xn)
  gemm_pw_kernel<0><<<dim3(144, 12, BATCH), blk, 0, stream>>>(
      wb_qkv, qkv_b, bufA, nullptr, bufD, 256, 768);
  // 6) depthwise on q/k/v branches
  dw_kernel<true, false><<<gDW, blk, 0, stream>>>(
      bufD + (size_t)0 * 256 * HW_, bufE + (size_t)0 * 256 * HW_,
      q_dw3_w, q_dw3_b, q_dw5_w, q_dw5_b, 768);
  dw_kernel<true, false><<<gDW, blk, 0, stream>>>(
      bufD + (size_t)1 * 256 * HW_, bufE + (size_t)1 * 256 * HW_,
      k_dw3_w, k_dw3_b, k_dw5_w, k_dw5_b, 768);
  dw_kernel<true, false><<<gDW, blk, 0, stream>>>(
      bufD + (size_t)2 * 256 * HW_, bufE + (size_t)2 * 256 * HW_,
      v_dw3_w, v_dw3_b, v_dw5_w, v_dw5_b, 768);
  // 7) linear attention
  attn_stats_kernel<<<dim3(BATCH * NH), blk, 0, stream>>>(bufE, kvbuf, ksbuf);
  attn_apply_kernel<<<gAP, blk, 0, stream>>>(bufE, kvbuf, ksbuf, bufC);
  // 8) sum = global_out(attn) + localf
  gemm_pw_kernel<1><<<dim3(144, 4, BATCH), blk, 0, stream>>>(
      wb_global, global_out_b, bufC, bufB, bufA, 256, 256);
  // 9) y1 = x + fuse_pw(sum)
  gemm_pw_kernel<2><<<dim3(144, 4, BATCH), blk, 0, stream>>>(
      wb_fuse, fuse_pw_b, bufA, x, y1, 256, 256);
  // 10) ffn norm
  ln_kernel<<<gLN, blk, 0, stream>>>(y1, ffn_g, ffn_b, bufB, Cc);
  // 11) ffn expand
  gemm_pw_kernel<0><<<dim3(144, 16, BATCH), blk, 0, stream>>>(
      wb_exp, ffn_exp_b, bufB, nullptr, bufF1, 256, HID);
  // 12) dw3 + exact gelu
  dw_kernel<false, true><<<dim3(HW_ / 256, HID, BATCH), blk, 0, stream>>>(
      bufF1, bufF2, ffn_dw3_w, ffn_dw3_b, nullptr, nullptr, HID);
  // 13) out = y1 + ffn_shrink(g)
  gemm_pw_kernel<2><<<dim3(144, 4, BATCH), blk, 0, stream>>>(
      wb_shr, ffn_shr_b, bufF2, y1, (float*)d_out, HID, 256);
}
